// Model_687194768042
// MI455X (gfx1250) — matrix operand order repaired and checked by reading, not compiled, not run on hardware
//
#include <hip/hip_runtime.h>
#include <hip/hip_bf16.h>

typedef __attribute__((ext_vector_type(16))) _Float16 v16h;
typedef __attribute__((ext_vector_type(8)))  float    v8f;

#define DIM        10000
#define T_STEPS    6
#define KDIM       1024          // 4 channels * 256 sample values
#define NCHUNKS    32            // KDIM / 32
#define NTILES     625           // 10000 / 16
#define NCLASSES   12
#define NGRID      40            // ceil(10000/256)

// ---------------------------------------------------------------------------
// Kernel 1: histogram of sample values -> counts[t][c][j]  (t<6, c<4, j<256)
// ---------------------------------------------------------------------------
__global__ void hd_count_kernel(const int* __restrict__ x, int* __restrict__ counts) {
    int b = blockIdx.x * blockDim.x + threadIdx.x;
    if (b >= 512) return;
    const int samples[T_STEPS] = {0, 51, 102, 153, 204, 255};
    #pragma unroll
    for (int t = 0; t < T_STEPS; ++t) {
        int s = samples[t];
        #pragma unroll
        for (int c = 0; c < 4; ++c) {
            int j = x[(b * 256 + s) * 4 + c] & 255;
            atomicAdd(&counts[t * KDIM + c * 256 + j], 1);
        }
    }
}

// ---------------------------------------------------------------------------
// Kernel 2: pre-swizzle counts into the WMMA A-fragment register layout.
// Apack[kchunk][lane][h] (f16, 32KB total) so each GEMM wave loads its v16h
// A operand as ONE contiguous 32B load (no predication, no cvt in hot loop).
// A-fragment layout (16x32 f16): lane carries row m=lane&15;
//   lanes 0-15 : halves 0-7 = K0..7,  halves 8-15 = K16..23
//   lanes 16-31: halves 0-7 = K8..15, halves 8-15 = K24..31
// Rows 6..15 are zero padding.
// ---------------------------------------------------------------------------
__global__ void hd_pack_a(const int* __restrict__ counts, _Float16* __restrict__ Apack) {
    int t = blockIdx.x * blockDim.x + threadIdx.x;   // 0 .. 1023
    if (t >= NCHUNKS * 32) return;
    int kc     = t >> 5;
    int lane   = t & 31;
    int m      = lane & 15;
    int hiHalf = lane >> 4;
    _Float16* dst = Apack + (kc * 32 + lane) * 16;
    #pragma unroll
    for (int h = 0; h < 16; ++h) {
        int K = h + ((h >= 8) ? 8 : 0) + hiHalf * 8;
        float av = (m < T_STEPS) ? (float)counts[m * KDIM + kc * 32 + K] : 0.0f;
        dst[h] = (_Float16)av;
    }
}

// ---------------------------------------------------------------------------
// Kernel 3: S(6x10000) = counts(6x1024) x P(1024x10000) via WMMA f16->f32.
// P[k][d] = iM[k,d]*CiM[c*21+min(j,20), d]  (note c*256+j == k), P in {+1,-1}.
// counts<=512 and +-1 are exact in f16; partial sums <=2048 exact in f32,
// so the WMMA GEMM is bit-exact. One wave per 16-column tile of D.
// ---------------------------------------------------------------------------
__global__ void __launch_bounds__(128)
hd_wmma_gemm(const v16h*  __restrict__ Apack,
             const float* __restrict__ iM,
             const float* __restrict__ CiM,
             float*       __restrict__ S) {
    const int wave   = threadIdx.x >> 5;
    const int lane   = threadIdx.x & 31;
    const int tile   = blockIdx.x * 4 + wave;
    if (tile >= NTILES) return;

    const int n      = lane & 15;     // column within tile (B/D lane role)
    const int hiHalf = lane >> 4;     // lane group 0/1
    const int d      = tile * 16 + n; // global column

    const float* iMd  = iM  + d;
    const float* CiMd = CiM + d;

    v8f acc = {};
    for (int kk = 0; kk < KDIM; kk += 32) {
        // --- A fragment: one contiguous pre-swizzled 32B load per lane.
        v16h a = Apack[(kk >> 5) * 32 + lane];

        // --- B fragment: 32x16 f16. lanes 0-15 hold K=0..7,16..23,
        // lanes 16-31 hold K=8..15,24..31 (same as A), column = lane & 15.
        v16h b;
        #pragma unroll
        for (int h = 0; h < 16; ++h) {
            int k = kk + h + ((h >= 8) ? 8 : 0) + hiHalf * 8;
            int c = k >> 8;
            int j = k & 255;
            int l = (j > 20) ? 20 : j;
            float p = iMd[k * DIM] * CiMd[(c * 21 + l) * DIM];
            b[h] = (_Float16)p;
        }
        acc = __builtin_amdgcn_wmma_f32_16x16x32_f16(
                  false, a, false, b, (short)0, acc, false, false);
    }
    // D layout: VGPR r, lanes 0-15 => row M=r, col N=lane. Rows 0..5 wanted.
    if (hiHalf == 0) {
        #pragma unroll
        for (int r = 0; r < T_STEPS; ++r) S[r * DIM + d] = acc[r];
    }
}

// ---------------------------------------------------------------------------
// N-gram kernels. roll(g,1)[d]=g[d-1] (wraparound), each g added to ALL rows:
//   g1[d] = S0[d-3]*S1[d-2]*S2[d-1]*S3[d]
//   g2[d] = (S1+g1)[d-3]*(S2+g1)[d-2]*(S3+g1)[d-1]*(S4+g1)[d]
//   g3[d] = (S2+g1+g2)[d-3]*...*(S5+g1+g2)[d]
//   final[d] = sum_t S[t][d] + 6*(g1+g2+g3)[d]
// Multiplication kept left-to-right to match reference rounding.
// ---------------------------------------------------------------------------
__device__ __forceinline__ int wrapm(int d, int off) { return (d - off + DIM) % DIM; }

__global__ void hd_ngram1(const float* __restrict__ S, float* __restrict__ G1) {
    int d = blockIdx.x * blockDim.x + threadIdx.x;
    if (d >= DIM) return;
    int w1 = wrapm(d,1), w2 = wrapm(d,2), w3 = wrapm(d,3);
    float g = S[0 * DIM + w3];
    g *= S[1 * DIM + w2];
    g *= S[2 * DIM + w1];
    g *= S[3 * DIM + d];
    G1[d] = g;
}

__global__ void hd_ngram2(const float* __restrict__ S, const float* __restrict__ G1,
                          float* __restrict__ G2) {
    int d = blockIdx.x * blockDim.x + threadIdx.x;
    if (d >= DIM) return;
    int w1 = wrapm(d,1), w2 = wrapm(d,2), w3 = wrapm(d,3);
    float g = S[1 * DIM + w3] + G1[w3];
    g *= S[2 * DIM + w2] + G1[w2];
    g *= S[3 * DIM + w1] + G1[w1];
    g *= S[4 * DIM + d]  + G1[d];
    G2[d] = g;
}

__global__ void hd_finalize(const float* __restrict__ S, const float* __restrict__ G1,
                            const float* __restrict__ G2, const float* __restrict__ W,
                            float* __restrict__ partial) {
    int d = blockIdx.x * blockDim.x + threadIdx.x;
    float enc = 0.0f;
    if (d < DIM) {
        int w1 = wrapm(d,1), w2 = wrapm(d,2), w3 = wrapm(d,3);
        float g3 = S[2 * DIM + w3] + G1[w3] + G2[w3];
        g3 *= S[3 * DIM + w2] + G1[w2] + G2[w2];
        g3 *= S[4 * DIM + w1] + G1[w1] + G2[w1];
        g3 *= S[5 * DIM + d]  + G1[d]  + G2[d];
        float fin = 0.0f;
        #pragma unroll
        for (int t = 0; t < T_STEPS; ++t) fin += S[t * DIM + d];
        fin += 6.0f * (G1[d] + G2[d] + g3);
        enc = (fin > 0.0f) ? 1.0f : -1.0f;
    }
    __shared__ float red[256];
    for (int cls = 0; cls < NCLASSES; ++cls) {
        float v = (d < DIM) ? enc * W[cls * DIM + d] : 0.0f;
        red[threadIdx.x] = v;
        __syncthreads();
        for (int off = 128; off > 0; off >>= 1) {
            if (threadIdx.x < off) red[threadIdx.x] += red[threadIdx.x + off];
            __syncthreads();
        }
        if (threadIdx.x == 0) partial[blockIdx.x * NCLASSES + cls] = red[0];
        __syncthreads();
    }
}

__global__ void hd_reduce_out(const float* __restrict__ partial, float* __restrict__ out) {
    int cls = threadIdx.x;
    if (cls < NCLASSES) {
        float s = 0.0f;
        for (int b = 0; b < NGRID; ++b) s += partial[b * NCLASSES + cls];
        out[cls] = s;
    }
}

// ---------------------------------------------------------------------------
// Host launcher. d_in: 0=x(int32 512x256x4), 1=iM(f32 4x256x10000),
// 2=CiM(f32 4x21x10000), 3=W(f32 12x10000). d_out: 12 f32.
// ---------------------------------------------------------------------------
extern "C" void kernel_launch(void* const* d_in, const int* in_sizes, int n_in,
                              void* d_out, int out_size, void* d_ws, size_t ws_size,
                              hipStream_t stream) {
    const int*   x   = (const int*)  d_in[0];
    const float* iM  = (const float*)d_in[1];
    const float* CiM = (const float*)d_in[2];
    const float* W   = (const float*)d_in[3];
    float* out = (float*)d_out;

    char* ws = (char*)d_ws;
    // Workspace layout (all offsets 32B aligned):
    int*       counts  = (int*)      (ws + 0);        //  6*1024*4      = 24576 B
    _Float16*  Apack   = (_Float16*) (ws + 24576);    //  32*32*16*2    = 32768 B
    float*     S       = (float*)    (ws + 57344);    //  6*10000*4     = 240000 B
    float*     G1      = (float*)    (ws + 297344);   //  40000 B
    float*     G2      = (float*)    (ws + 337344);   //  40000 B
    float*     partial = (float*)    (ws + 377344);   //  40*12*4       = 1920 B

    hipMemsetAsync(counts, 0, T_STEPS * KDIM * sizeof(int), stream);

    hd_count_kernel<<<2, 256, 0, stream>>>(x, counts);
    hd_pack_a<<<4, 256, 0, stream>>>(counts, Apack);

    // 625 column tiles, 4 waves (tiles) per 128-thread block.
    hd_wmma_gemm<<<(NTILES + 3) / 4, 128, 0, stream>>>((const v16h*)Apack, iM, CiM, S);

    hd_ngram1<<<NGRID, 256, 0, stream>>>(S, G1);
    hd_ngram2<<<NGRID, 256, 0, stream>>>(S, G1, G2);
    hd_finalize<<<NGRID, 256, 0, stream>>>(S, G1, G2, W, partial);
    hd_reduce_out<<<1, 32, 0, stream>>>(partial, out);
}
